// DepGCN_75780402970819
// MI455X (gfx1250) — matrix operand order repaired and checked by reading, not compiled, not run on hardware
//
#include <hip/hip_runtime.h>
#include <stdint.h>

// ---------------------------------------------------------------------------
// Two-layer GCN forward for MI455X (gfx1250, wave32).
//  - GEMMs via V_WMMA_F32_16X16X32_BF16 (bf16 operands, f32 accumulate)
//    with LDS-staged B tile (conflict-free pitch) + double-buffered A frags
//  - Edge aggregation: per-edge wave gather + global f32 atomic scatter-add
//  - Fused self-loop term in GEMM epilogue; layer-2 accumulator in d_out
// ---------------------------------------------------------------------------

#define DM 512           // d_model == hidden == 512 (fixed by reference)
typedef __attribute__((ext_vector_type(16))) __bf16 v16bf;
typedef __attribute__((ext_vector_type(8)))  float  v8f;

union FragBF { uint4 u[2]; v16bf v; };

__device__ __forceinline__ unsigned short f2bf(float f) {
    unsigned int u = __float_as_uint(f);
    unsigned int r = (u + 0x7FFFu + ((u >> 16) & 1u)) >> 16;   // RNE
    return (unsigned short)r;
}

// ---------------- small utility kernels ----------------

__global__ void k_fill_f32(float* __restrict__ p, float v, int n) {
    int i = blockIdx.x * blockDim.x + threadIdx.x;
    if (i < n) p[i] = v;
}

__global__ void k_deg_edges(const int* __restrict__ dst, float* __restrict__ deg, int nE) {
    int e = blockIdx.x * blockDim.x + threadIdx.x;
    if (e < nE) atomicAdd(&deg[dst[e]], 1.0f);
}

__global__ void k_inv_sqrt(const float* __restrict__ deg, float* __restrict__ inv, int n) {
    int i = blockIdx.x * blockDim.x + threadIdx.x;
    if (i < n) inv[i] = rsqrtf(deg[i]);          // deg >= 1 (self loop)
}

// x (f32, row-major NxD) -> bf16, zero-filled pad rows
__global__ void k_to_bf16_pad(const float* __restrict__ x, unsigned short* __restrict__ xb,
                              int nRows, int nRowsPad) {
    int i = blockIdx.x * blockDim.x + threadIdx.x;
    if (i >= nRowsPad * DM) return;
    int row = i >> 9;                            // /512
    xb[i] = (row < nRows) ? f2bf(x[i]) : (unsigned short)0;
}

// W (f32, K x N row-major) -> Wt (bf16, N x K row-major)
__global__ void k_w_transpose_bf16(const float* __restrict__ W, unsigned short* __restrict__ Wt) {
    int i = blockIdx.x * blockDim.x + threadIdx.x;   // i = n*512 + k
    if (i >= DM * DM) return;
    int n = i >> 9, k = i & (DM - 1);
    Wt[i] = f2bf(W[k * DM + n]);
}

// ---------------- bf16 WMMA GEMM: H = A(bf16) x B(bf16), AggInit = H*inv^2 --
// A:  [Mpad x 512] bf16 row-major.  Bt: [512 x 512] bf16, Bt[n][k] = W[k][n].
// Block = 256 threads = 8 waves; block tile = 128 rows x 64 cols.
// B tile staged in LDS in two K-halves (pitch 264 -> ds bank-conflict-free);
// A fragments double-buffered in registers across the unrolled K loop.

#define KH     256                // K elements per LDS stage
#define BPITCH 264                // LDS row pitch (elems): 132 dwords, %64 == 4

__global__ __launch_bounds__(256) void k_gemm_bf16_wmma(
    const unsigned short* __restrict__ A,
    const unsigned short* __restrict__ Bt,
    const float* __restrict__ inv,
    float* __restrict__ H,
    float* __restrict__ AggInit,
    int M)
{
    __shared__ unsigned short Bs[64 * BPITCH];   // 33,792 B

    const int lane  = threadIdx.x & 31;
    const int wave  = threadIdx.x >> 5;
    const int mbase = blockIdx.x * 128 + wave * 16;
    const int nbase = blockIdx.y * 64;
    const bool hi   = lane >= 16;
    const int  l15  = lane & 15;

    // A fragment (16x32 bf16): lane row = mbase + l15; hi half shifts K by +16.
    const unsigned short* arow = A + (size_t)(mbase + l15) * DM + (hi ? 16 : 0);
    // B fragment base in LDS: lane col-row = l15; hi half covers K+16..31.
    const unsigned short* bl = Bs + l15 * BPITCH + (hi ? 16 : 0);

    v8f acc0 = {}, acc1 = {}, acc2 = {}, acc3 = {};

    for (int half = 0; half < 2; ++half) {
        const int kbase = half * KH;

        // ---- cooperative stage of B tile half: 64 rows x 256 elems ----
        // 2048 uint4 chunks, 8 per thread, fully coalesced 16B stores to LDS.
        for (int chunk = threadIdx.x; chunk < 2048; chunk += 256) {
            int c  = chunk >> 5;                 // row (output column), 0..63
            int kc = (chunk & 31) * 8;           // elem offset within row
            *(uint4*)(Bs + c * BPITCH + kc) =
                *(const uint4*)(Bt + (size_t)(nbase + c) * DM + kbase + kc);
        }
        __syncthreads();

        // ---- K loop over this half, A double-buffered ----
        FragBF a_cur;
        a_cur.u[0] = *(const uint4*)(arow + kbase);
        a_cur.u[1] = *(const uint4*)(arow + kbase + 8);
#pragma unroll
        for (int t = 0; t < KH / 32; ++t) {
            const int k = t * 32;
            FragBF a_nxt;
            if (t < KH / 32 - 1) {               // prefetch next A fragment
                a_nxt.u[0] = *(const uint4*)(arow + kbase + k + 32);
                a_nxt.u[1] = *(const uint4*)(arow + kbase + k + 40);
            }
            FragBF b;
            b.u[0] = *(const uint4*)(bl + k);
            b.u[1] = *(const uint4*)(bl + k + 8);
            acc0 = __builtin_amdgcn_wmma_f32_16x16x32_bf16(false, a_cur.v, false, b.v,
                                                           (short)0, acc0, false, false);
            b.u[0] = *(const uint4*)(bl + 16 * BPITCH + k);
            b.u[1] = *(const uint4*)(bl + 16 * BPITCH + k + 8);
            acc1 = __builtin_amdgcn_wmma_f32_16x16x32_bf16(false, a_cur.v, false, b.v,
                                                           (short)0, acc1, false, false);
            b.u[0] = *(const uint4*)(bl + 32 * BPITCH + k);
            b.u[1] = *(const uint4*)(bl + 32 * BPITCH + k + 8);
            acc2 = __builtin_amdgcn_wmma_f32_16x16x32_bf16(false, a_cur.v, false, b.v,
                                                           (short)0, acc2, false, false);
            b.u[0] = *(const uint4*)(bl + 48 * BPITCH + k);
            b.u[1] = *(const uint4*)(bl + 48 * BPITCH + k + 8);
            acc3 = __builtin_amdgcn_wmma_f32_16x16x32_bf16(false, a_cur.v, false, b.v,
                                                           (short)0, acc3, false, false);
            if (t < KH / 32 - 1) a_cur = a_nxt;
        }
        __syncthreads();                         // before next half overwrites Bs
    }

    // C/D layout: VGPR r -> (M = r + (hi?8:0), N = l15)
    const int rowoff = hi ? 8 : 0;
#pragma unroll
    for (int r = 0; r < 8; ++r) {
        int row = mbase + rowoff + r;
        if (row < M) {
            float s  = inv[row];
            float s2 = s * s;                    // self-loop norm 1/deg
            size_t base = (size_t)row * DM + nbase + l15;
            H[base +  0] = acc0[r];  AggInit[base +  0] = acc0[r] * s2;
            H[base + 16] = acc1[r];  AggInit[base + 16] = acc1[r] * s2;
            H[base + 32] = acc2[r];  AggInit[base + 32] = acc2[r] * s2;
            H[base + 48] = acc3[r];  AggInit[base + 48] = acc3[r] * s2;
        }
    }
}

// ---------------- edge aggregation: agg[dst] += h[src] * inv[s]*inv[d] ------
// One wave per edge; lanes stride features in float4 chunks (coalesced 512B).

__global__ __launch_bounds__(256) void k_edge_agg(
    const int* __restrict__ src, const int* __restrict__ dst,
    const float* __restrict__ inv,
    const float* __restrict__ h, float* __restrict__ agg, int nE)
{
    int e = blockIdx.x * 8 + (threadIdx.x >> 5);
    if (e >= nE) return;                           // wave-uniform exit
    int lane = threadIdx.x & 31;
    int s = src[e], d = dst[e];
    float norm = inv[s] * inv[d];
    const float4* hs = (const float4*)(h + (size_t)s * DM);
    float* ad = agg + (size_t)d * DM;
    __builtin_prefetch(hs, 0, 1);                  // global_prefetch_b8
#pragma unroll
    for (int i = 0; i < 4; ++i) {
        int c = lane + 32 * i;                     // float4 index, 0..127
        float4 v = hs[c];
        int f = c * 4;
        atomicAdd(ad + f + 0, v.x * norm);
        atomicAdd(ad + f + 1, v.y * norm);
        atomicAdd(ad + f + 2, v.z * norm);
        atomicAdd(ad + f + 3, v.w * norm);
    }
}

// ---------------- epilogues ----------------

// x1b = bf16(relu(agg + b)), padded rows -> 0
__global__ void k_bias_relu_bf16(const float* __restrict__ agg, const float* __restrict__ b,
                                 unsigned short* __restrict__ out, int nRows, int nRowsPad) {
    int i = blockIdx.x * blockDim.x + threadIdx.x;
    if (i >= nRowsPad * DM) return;
    int row = i >> 9;
    if (row < nRows) {
        float v = agg[i] + b[i & (DM - 1)];
        out[i] = f2bf(v > 0.0f ? v : 0.0f);
    } else {
        out[i] = 0;
    }
}

// in-place: x2 = relu(x2 + b)
__global__ void k_bias_relu_f32(float* __restrict__ x2, const float* __restrict__ b, int n) {
    int i = blockIdx.x * blockDim.x + threadIdx.x;
    if (i < n) {
        float v = x2[i] + b[i & (DM - 1)];
        x2[i] = v > 0.0f ? v : 0.0f;
    }
}

// per-feature max over nodes; 512 threads/block, feature = threadIdx.x.
// Values are >= 0 (post-ReLU) so uint atomicMax on float bits is order-correct.
__global__ __launch_bounds__(512) void k_colmax(const float* __restrict__ x2,
                                                float* __restrict__ p, int nRows) {
    int f = threadIdx.x;
    float m = 0.0f;
    for (int row = blockIdx.x; row < nRows; row += gridDim.x)
        m = fmaxf(m, x2[(size_t)row * DM + f]);
    atomicMax((unsigned int*)(p + f), __float_as_uint(m));
}

// ---------------------------------------------------------------------------

extern "C" void kernel_launch(void* const* d_in, const int* in_sizes, int n_in,
                              void* d_out, int out_size, void* d_ws, size_t ws_size,
                              hipStream_t stream) {
    (void)n_in; (void)out_size; (void)ws_size;

    const float* x  = (const float*)d_in[0];
    const int*   ei = (const int*)d_in[1];     // [2, E] int32 (JAX x64 off)
    const float* W1 = (const float*)d_in[2];
    const float* b1 = (const float*)d_in[3];
    const float* W2 = (const float*)d_in[4];
    const float* b2 = (const float*)d_in[5];

    const int N = in_sizes[0] / DM;            // 20000 nodes
    const int E = in_sizes[1] / 2;             // 320000 edges
    const int NPAD = ((N + 127) / 128) * 128;  // 20096
    const int* src = ei;
    const int* dst = ei + E;

    float* x2 = (float*)d_out;                 // [N x 512]
    float* p  = x2 + (size_t)N * DM;           // [512]

    // ---- carve workspace (all 256B-aligned) ----
    char* w = (char*)d_ws;
    size_t off = 0;
    auto carve = [&](size_t bytes) { void* r = w + off; off += (bytes + 255) & ~(size_t)255; return r; };
    float*          deg  = (float*)         carve((size_t)N * 4);
    float*          invd = (float*)         carve((size_t)N * 4);
    unsigned short* Wt1  = (unsigned short*)carve((size_t)DM * DM * 2);
    unsigned short* Wt2  = (unsigned short*)carve((size_t)DM * DM * 2);
    unsigned short* xb   = (unsigned short*)carve((size_t)NPAD * DM * 2);
    unsigned short* x1b  = (unsigned short*)carve((size_t)NPAD * DM * 2);
    float*          h    = (float*)         carve((size_t)N * DM * 4);   // shared L1/L2
    float*          agg1 = (float*)         carve((size_t)N * DM * 4);

    const int T = 256;
    // degrees (self-loop contributes 1), symmetric norm
    k_fill_f32<<<(N + T - 1) / T, T, 0, stream>>>(deg, 1.0f, N);
    k_deg_edges<<<(E + T - 1) / T, T, 0, stream>>>(dst, deg, E);
    k_inv_sqrt<<<(N + T - 1) / T, T, 0, stream>>>(deg, invd, N);

    // bf16 operands
    k_to_bf16_pad<<<((size_t)NPAD * DM + T - 1) / T, T, 0, stream>>>(x, xb, N, NPAD);
    k_w_transpose_bf16<<<(DM * DM + T - 1) / T, T, 0, stream>>>(W1, Wt1);
    k_w_transpose_bf16<<<(DM * DM + T - 1) / T, T, 0, stream>>>(W2, Wt2);

    dim3 ggrid(NPAD / 128, DM / 64);

    // ---- layer 1 ----
    k_gemm_bf16_wmma<<<ggrid, 256, 0, stream>>>(xb, Wt1, invd, h, agg1, N);
    k_edge_agg<<<(E + 7) / 8, 256, 0, stream>>>(src, dst, invd, h, agg1, E);
    k_bias_relu_bf16<<<((size_t)NPAD * DM + T - 1) / T, T, 0, stream>>>(agg1, b1, x1b, N, NPAD);

    // ---- layer 2 (accumulator lives in d_out) ----
    k_gemm_bf16_wmma<<<ggrid, 256, 0, stream>>>(x1b, Wt2, invd, h, x2, N);
    k_edge_agg<<<(E + 7) / 8, 256, 0, stream>>>(src, dst, invd, h, x2, E);
    k_bias_relu_f32<<<((size_t)N * DM + T - 1) / T, T, 0, stream>>>(x2, b2, N * DM);

    // ---- per-feature max ----
    k_fill_f32<<<(DM + T - 1) / T, T, 0, stream>>>(p, 0.0f, DM);
    k_colmax<<<160, DM, 0, stream>>>(x2, p, N);
}
